// HConstructor20_69363721830613
// MI455X (gfx1250) — matrix operand order repaired and checked by reading, not compiled, not run on hardware
//
#include <hip/hip_runtime.h>
#include <math.h>

#define N_NODES 50000
#define N_EDGES 800000
#define DIM     256
#define T_REP   4
#define NS_CL   64
#define NT_NODES (N_NODES * (T_REP + 1))                       // 250000
#define SEG_C    (2 * N_EDGES)                                 // start of extra edges
#define SEG_D    (2 * N_EDGES + T_REP * N_NODES)               // start of self loops
#define EN_EDGES (2 * N_EDGES + T_REP * N_NODES + NT_NODES)    // 2,050,000

typedef __attribute__((ext_vector_type(16))) _Float16 v16h;
typedef __attribute__((ext_vector_type(8)))  _Float16 v8h;
typedef __attribute__((ext_vector_type(8)))  float    v8f;

__device__ __forceinline__ float wred_sum(float v) {
#pragma unroll
  for (int off = 16; off > 0; off >>= 1) v += __shfl_xor(v, off, 32);
  return v;
}

// ---------------------------------------------------------------------------
// Y[M x Ncols] = act(X[M x 256]) @ W[256 x Ncols] * outScale + bias
// wTrans: W element (k,n) read as W[n*256 + k] (for hyperedge_features.T).
// Block 256 threads (8 waves); tile 128 rows x 64 cols; wave = 16 rows x 64 cols.
// ---------------------------------------------------------------------------
__global__ __launch_bounds__(256)
void wmma_gemm_kernel(const float* __restrict__ X, const float* __restrict__ W,
                      const float* __restrict__ bias, float* __restrict__ Y,
                      int M, int Ncols, int wTrans, int reluIn, float outScale)
{
  __shared__ __align__(64) _Float16 sX[128][32];  // 8 KB
  __shared__ __align__(64) _Float16 sW[64][32];   // 4 KB, transposed [n][k]
  const int tid  = threadIdx.x;
  const int wave = tid >> 5;
  const int lane = tid & 31;
  const int row0 = blockIdx.x * 128;
  const int col0 = blockIdx.y * 64;

  v8f acc[4] = {};

  for (int k0 = 0; k0 < DIM; k0 += 32) {
    { // stage X tile, f32 -> f16, fused input relu
      const int r  = tid >> 1;
      const int cs = (tid & 1) << 4;
      const int gr = row0 + r;
      float v[16];
      if (gr < M) {
        const float* p = X + (size_t)gr * DIM + k0 + cs;
#pragma unroll
        for (int j = 0; j < 16; ++j) v[j] = p[j];
      } else {
#pragma unroll
        for (int j = 0; j < 16; ++j) v[j] = 0.0f;
      }
      if (reluIn) {
#pragma unroll
        for (int j = 0; j < 16; ++j) v[j] = fmaxf(v[j], 0.0f);
      }
#pragma unroll
      for (int j = 0; j < 16; ++j) sX[r][cs + j] = (_Float16)v[j];
    }
    { // stage W tile transposed into sW[n][k]
      const int k  = tid >> 3;
      const int ns = (tid & 7) << 3;
#pragma unroll
      for (int j = 0; j < 8; ++j) {
        float wv = wTrans ? W[(size_t)(col0 + ns + j) * DIM + (k0 + k)]
                          : W[(size_t)(k0 + k) * Ncols + (col0 + ns + j)];
        sW[ns + j][k] = (_Float16)wv;
      }
    }
    __syncthreads();

    // A fragment: lanes 0-15 row m, K halves {0..7,16..23}; lanes 16-31 {8..15,24..31}
    const int mrow = (wave << 4) + (lane & 15);
    const int klo  = (lane < 16) ? 0 : 8;
    v8h alo = *(const v8h*)&sX[mrow][klo];
    v8h ahi = *(const v8h*)&sX[mrow][16 + klo];
    v16h afrag;
#pragma unroll
    for (int j = 0; j < 8; ++j) { afrag[j] = alo[j]; afrag[8 + j] = ahi[j]; }

    const int ks = (lane < 16) ? 0 : 8;
#pragma unroll
    for (int nt = 0; nt < 4; ++nt) {
      const int c = (nt << 4) + (lane & 15);
      v8h blo = *(const v8h*)&sW[c][ks];
      v8h bhi = *(const v8h*)&sW[c][16 + ks];
      v16h bfrag;
#pragma unroll
      for (int j = 0; j < 8; ++j) { bfrag[j] = blo[j]; bfrag[8 + j] = bhi[j]; }
      acc[nt] = __builtin_amdgcn_wmma_f32_16x16x32_f16(
          false, afrag, false, bfrag, (short)0, acc[nt], false, false);
    }
    __syncthreads();
  }

  // C/D layout: lane<16 -> col=lane, rows r..; lane>=16 -> col=lane-16, rows r+8
  const int rbase = row0 + (wave << 4) + ((lane < 16) ? 0 : 8);
#pragma unroll
  for (int nt = 0; nt < 4; ++nt) {
    const int c  = col0 + (nt << 4) + (lane & 15);
    const float bv = bias ? bias[c] : 0.0f;
#pragma unroll
    for (int r = 0; r < 8; ++r) {
      const int gr = rbase + r;
      if (gr < M) Y[(size_t)gr * Ncols + c] = acc[nt][r] * outScale + bv;
    }
  }
}

// ---------------------------------------------------------------------------
__global__ void zero_kernel(float* p, size_t n) {
  size_t i = (size_t)blockIdx.x * blockDim.x + threadIdx.x;
  if (i < n) p[i] = 0.0f;
}

__global__ void initbias_kernel(float* out, const float* __restrict__ b, size_t n) {
  size_t i = (size_t)blockIdx.x * blockDim.x + threadIdx.x;
  if (i < n) out[i] = b[i & (DIM - 1)];
}

// inverse L2 norm per row (wave per row)
__global__ void rnorm_kernel(const float* __restrict__ AF, float* __restrict__ rnorm, int rows) {
  int r = blockIdx.x * (blockDim.x >> 5) + (threadIdx.x >> 5);
  if (r >= rows) return;
  int lane = threadIdx.x & 31;
  const float* p = AF + (size_t)r * DIM + lane * 8;
  float s = 0.0f;
#pragma unroll
  for (int j = 0; j < 8; ++j) { float v = p[j]; s += v * v; }
  s = wred_sum(s);
  if (lane == 0) rnorm[r] = 1.0f / fmaxf(sqrtf(s), 1e-8f);
}

// per-edge best replica by cosine sim; writes segment B of augmented edge list
__global__ void sim_kernel(const int* __restrict__ src, const int* __restrict__ dst,
                           const float* __restrict__ AF, const float* __restrict__ rnorm,
                           int* __restrict__ esrc, int* __restrict__ edst) {
  int e = blockIdx.x * (blockDim.x >> 5) + (threadIdx.x >> 5);
  if (e >= N_EDGES) return;
  int lane = threadIdx.x & 31;
  int s = src[e], d0 = dst[e];
  const float* ps = AF + (size_t)s * DIM + lane * 8;
  float fv[8];
#pragma unroll
  for (int j = 0; j < 8; ++j) fv[j] = ps[j];
  float sims[T_REP];
#pragma unroll
  for (int t = 0; t < T_REP; ++t) {
    const float* pd = AF + ((size_t)(t + 1) * N_NODES + d0) * DIM + lane * 8;
    float p = 0.0f;
#pragma unroll
    for (int j = 0; j < 8; ++j) p += fv[j] * pd[j];
    sims[t] = wred_sum(p);
  }
  if (lane == 0) {
    float rs = rnorm[s];
    int best = 0; float bv = -1e30f;
#pragma unroll
    for (int t = 0; t < T_REP; ++t) {
      float v = sims[t] * rs * rnorm[(size_t)(t + 1) * N_NODES + d0];
      if (v > bv) { bv = v; best = t; }   // strict > : first max wins (jnp.argmax)
    }
    esrc[N_EDGES + e] = s;
    edst[N_EDGES + e] = d0 + best * N_NODES;
  }
}

// segments A (orig edges), C (replica identity edges), D (self loops)
__global__ void edges_fill_kernel(const int* __restrict__ src, const int* __restrict__ dst,
                                  int* __restrict__ esrc, int* __restrict__ edst) {
  int i = blockIdx.x * blockDim.x + threadIdx.x;
  if (i < N_EDGES) { esrc[i] = src[i]; edst[i] = dst[i]; }
  if (i < T_REP * N_NODES) {
    int rep = i / N_NODES, n = i - rep * N_NODES;
    esrc[SEG_C + i] = n;
    edst[SEG_C + i] = n + rep * N_NODES;
  }
  if (i < NT_NODES) { esrc[SEG_D + i] = i; edst[SEG_D + i] = i; }
}

// w2 (drop pre-existing self loops) + degree accumulation
__global__ void wdeg_kernel(const int* __restrict__ esrc, const int* __restrict__ edst,
                            float* __restrict__ enorm, float* __restrict__ deg) {
  int i = blockIdx.x * blockDim.x + threadIdx.x;
  if (i >= EN_EDGES) return;
  int s = esrc[i], d = edst[i];
  float w = (i >= SEG_D) ? 1.0f : ((s == d) ? 0.0f : 1.0f);
  enorm[i] = w;
  if (w != 0.0f) atomicAdd(&deg[d], w);
}

__global__ void dinv_kernel(float* deg) {
  int i = blockIdx.x * blockDim.x + threadIdx.x;
  if (i >= NT_NODES) return;
  float v = deg[i];
  deg[i] = (v > 0.0f) ? rsqrtf(v) : 0.0f;
}

__global__ void normfin_kernel(const int* __restrict__ esrc, const int* __restrict__ edst,
                               float* __restrict__ enorm, const float* __restrict__ dinv) {
  int i = blockIdx.x * blockDim.x + threadIdx.x;
  if (i >= EN_EDGES) return;
  enorm[i] *= dinv[esrc[i]] * dinv[edst[i]];
}

// out[dst] += norm * xw[src]   (wave per edge, 8 floats per lane)
__global__ void scatter_kernel(const int* __restrict__ esrc, const int* __restrict__ edst,
                               const float* __restrict__ enorm,
                               const float* __restrict__ xw, float* __restrict__ out) {
  int e = blockIdx.x * (blockDim.x >> 5) + (threadIdx.x >> 5);
  if (e >= EN_EDGES) return;
  float w = enorm[e];
  if (w == 0.0f) return;
  int lane = threadIdx.x & 31;
  int s = esrc[e], d = edst[e];
  const float* xp = xw + (size_t)s * DIM + lane * 8;
  float* op = out + (size_t)d * DIM + lane * 8;
#pragma unroll
  for (int j = 0; j < 8; ++j) atomicAdd(&op[j], w * xp[j]);
}

// class argmax per (replica, node) row; count into H[node][class]
__global__ void argmax_h_kernel(const float* __restrict__ logits, float* __restrict__ H) {
  int r = blockIdx.x * (blockDim.x >> 5) + (threadIdx.x >> 5);
  if (r >= NT_NODES) return;
  int lane = threadIdx.x & 31;
  float v1 = logits[(size_t)r * NS_CL + lane];
  float v2 = logits[(size_t)r * NS_CL + lane + 32];
  float v; int c;
  if (v2 > v1) { v = v2; c = lane + 32; } else { v = v1; c = lane; }
#pragma unroll
  for (int off = 16; off > 0; off >>= 1) {
    float ov = __shfl_xor(v, off, 32);
    int   oc = __shfl_xor(c, off, 32);
    if (ov > v || (ov == v && oc < c)) { v = ov; c = oc; }
  }
  if (lane == 0) atomicAdd(&H[(size_t)(r % N_NODES) * NS_CL + c], 1.0f);
}

// hyperedge_features[c] += AF2[n] for every c with H[n][c] > 0 (wave per node)
__global__ void hyper_kernel(const float* __restrict__ AF2, const float* __restrict__ H,
                             float* __restrict__ hyper) {
  int n = blockIdx.x * (blockDim.x >> 5) + (threadIdx.x >> 5);
  if (n >= N_NODES) return;
  int lane = threadIdx.x & 31;
  const float* p = AF2 + (size_t)n * DIM + lane * 8;
  float a[8];
#pragma unroll
  for (int j = 0; j < 8; ++j) a[j] = p[j];
  for (int c = 0; c < NS_CL; ++c) {
    if (H[(size_t)n * NS_CL + c] > 0.0f) {
      float* hp = hyper + (size_t)c * DIM + lane * 8;
#pragma unroll
      for (int j = 0; j < 8; ++j) atomicAdd(&hp[j], a[j]);
    }
  }
}

// ---------------------------------------------------------------------------
extern "C" void kernel_launch(void* const* d_in, const int* in_sizes, int n_in,
                              void* d_out, int out_size, void* d_ws, size_t ws_size,
                              hipStream_t stream) {
  const int*   edge_index = (const int*)d_in[0];
  const float* features   = (const float*)d_in[1];
  const float* lin_w      = (const float*)d_in[2];
  const float* lin_b      = (const float*)d_in[3];
  const float* gcn0_w     = (const float*)d_in[4];
  const float* gcn0_b     = (const float*)d_in[5];
  const float* gcn1_w     = (const float*)d_in[6];
  const float* gcn1_b     = (const float*)d_in[7];
  const float* lin1_w     = (const float*)d_in[8];
  const float* lin1_b     = (const float*)d_in[9];
  const int* src = edge_index;
  const int* dst = edge_index + N_EDGES;

  char* ws = (char*)d_ws;
  size_t off = 0;
  auto carve = [&](size_t bytes) -> void* {
    void* p = ws + off;
    off += (bytes + 255) & ~(size_t)255;
    return p;
  };
  const size_t BIG = (size_t)NT_NODES * DIM;            // 64,000,000 floats
  float* AF    = (float*)carve(BIG * 4);                // all_features
  float* B1    = (float*)carve(BIG * 4);                // xw scratch / logits
  float* B2    = (float*)carve(BIG * 4);                // h
  float* AF2   = (float*)carve(BIG * 4);                // all_features2
  float* rnorm = (float*)carve((size_t)NT_NODES * 4);
  float* deg   = (float*)carve((size_t)NT_NODES * 4);   // becomes dinv in place
  float* enorm = (float*)carve((size_t)EN_EDGES * 4);
  int*   esrc  = (int*)carve((size_t)EN_EDGES * 4);
  int*   edst  = (int*)carve((size_t)EN_EDGES * 4);

  float* H_out = (float*)d_out;                                   // (N, NS)
  float* hyper = H_out + (size_t)N_NODES * NS_CL;                 // (NS, D)
  float* dots  = hyper + (size_t)NS_CL * DIM;                     // (NT, NS)

  dim3 blk(256);

  // all_features rows [0,N) = features
  hipMemcpyAsync(AF, features, (size_t)N_NODES * DIM * sizeof(float),
                 hipMemcpyDeviceToDevice, stream);

  // transformed[t] = features @ lin_w[t] + lin_b[t]  -> AF rows [(1+t)N, (2+t)N)
  {
    dim3 g((N_NODES + 127) / 128, DIM / 64);
    for (int t = 0; t < T_REP; ++t)
      wmma_gemm_kernel<<<g, blk, 0, stream>>>(
          features, lin_w + (size_t)t * DIM * DIM, lin_b + (size_t)t * DIM,
          AF + (size_t)(t + 1) * N_NODES * DIM, N_NODES, DIM, 0, 0, 1.0f);
  }

  rnorm_kernel<<<(NT_NODES + 7) / 8, blk, 0, stream>>>(AF, rnorm, NT_NODES);
  sim_kernel<<<(N_EDGES + 7) / 8, blk, 0, stream>>>(src, dst, AF, rnorm, esrc, edst);
  edges_fill_kernel<<<(N_EDGES + 255) / 256, blk, 0, stream>>>(src, dst, esrc, edst);
  zero_kernel<<<(NT_NODES + 255) / 256, blk, 0, stream>>>(deg, (size_t)NT_NODES);
  wdeg_kernel<<<(EN_EDGES + 255) / 256, blk, 0, stream>>>(esrc, edst, enorm, deg);
  dinv_kernel<<<(NT_NODES + 255) / 256, blk, 0, stream>>>(deg);
  normfin_kernel<<<(EN_EDGES + 255) / 256, blk, 0, stream>>>(esrc, edst, enorm, deg);

  // gcn0: h = scatter(norm * (relu(AF)@W0)[src]) + b0
  {
    dim3 g((NT_NODES + 127) / 128, DIM / 64);
    wmma_gemm_kernel<<<g, blk, 0, stream>>>(AF, gcn0_w, nullptr, B1, NT_NODES, DIM, 0, 1, 1.0f);
  }
  initbias_kernel<<<(int)((BIG + 255) / 256), blk, 0, stream>>>(B2, gcn0_b, BIG);
  scatter_kernel<<<(EN_EDGES + 7) / 8, blk, 0, stream>>>(esrc, edst, enorm, B1, B2);

  // gcn1: AF2 = scatter(norm * (relu(h)@W1)[src]) + b1
  {
    dim3 g((NT_NODES + 127) / 128, DIM / 64);
    wmma_gemm_kernel<<<g, blk, 0, stream>>>(B2, gcn1_w, nullptr, B1, NT_NODES, DIM, 0, 1, 1.0f);
  }
  initbias_kernel<<<(int)((BIG + 255) / 256), blk, 0, stream>>>(AF2, gcn1_b, BIG);
  scatter_kernel<<<(EN_EDGES + 7) / 8, blk, 0, stream>>>(esrc, edst, enorm, B1, AF2);

  // logits = relu(AF2) @ lin1_w + lin1_b  (into B1, stride NS)
  {
    dim3 g((NT_NODES + 127) / 128, 1);
    wmma_gemm_kernel<<<g, blk, 0, stream>>>(AF2, lin1_w, lin1_b, B1, NT_NODES, NS_CL, 0, 1, 1.0f);
  }

  // H: per-node class counts over (t+1) replicas
  zero_kernel<<<(int)(((size_t)N_NODES * NS_CL + 255) / 256), blk, 0, stream>>>(
      H_out, (size_t)N_NODES * NS_CL);
  argmax_h_kernel<<<(NT_NODES + 7) / 8, blk, 0, stream>>>(B1, H_out);

  // hyperedge_features = (H>0).T @ AF2[:N]
  zero_kernel<<<(NS_CL * DIM + 255) / 256, blk, 0, stream>>>(hyper, (size_t)NS_CL * DIM);
  hyper_kernel<<<(N_NODES + 7) / 8, blk, 0, stream>>>(AF2, H_out, hyper);

  // dots = AF @ hyperedge_features.T * D^-0.5
  {
    dim3 g((NT_NODES + 127) / 128, 1);
    wmma_gemm_kernel<<<g, blk, 0, stream>>>(AF, hyper, nullptr, dots, NT_NODES, NS_CL, 1, 0, 0.0625f);
  }
}
